// InstantNeRF_20899310862906
// MI455X (gfx1250) — compile-verified, matrix operand order checked by reading, not run on hardware
//
#include <hip/hip_runtime.h>

// ---------------------------------------------------------------------------
// Instant-NGP forward: hash-grid encoding + 32->64->64->64->16 MLP, fused.
// CDNA5 (gfx1250, wave32): MLP on v_wmma_f32_16x16x32_f16; weight staging via
// GLOBAL_LOAD_ASYNC_TO_LDS_B128 (overlapped with the hash-gather phase).
// ---------------------------------------------------------------------------

#define NUM_LEVEL   16
#define LOG_T       19
#define TABLE_SIZE  (1u << LOG_T)
#define N_POINTS    262144
#define PTS_PER_BLK 128

// fp16 weight layout inside LDS / workspace (halves):
//   [0,2048)     w_in  (64x32)
//   [2048,6144)  w_h0  (64x64)
//   [6144,10240) w_h1  (64x64)
//   [10240,11264)w_out (16x64)
#define W_TOTAL   11264
#define W_IN_OFF  0
#define W_H0_OFF  2048
#define W_H1_OFF  6144
#define W_OUT_OFF 10240
#define B_TOTAL   208     // 64+64+64+16 f32 biases

typedef _Float16 v16h __attribute__((ext_vector_type(16)));
typedef _Float16 v8h  __attribute__((ext_vector_type(8)));
typedef float    v8f  __attribute__((ext_vector_type(8)));
typedef int      v4i  __attribute__((ext_vector_type(4)));

#if __has_builtin(__builtin_amdgcn_global_load_async_to_lds_b128)
#define USE_ASYNC_LDS 1
#else
#define USE_ASYNC_LDS 0
#endif

// The async-to-LDS builtins take generic int4* pointers (pointee encodes the
// 16B transfer size). Integer detour strips const and any addrspace sugar.
#define V4IP(p) ((v4i*)(unsigned long long)(p))

// Load one 32-wide-K half-fragment slice for WMMA (A or B operand).
// Layout per ISA 7.12.2 (16-bit A 16x32): lane half h covers K = h*8..h*8+7
// and K = 16+h*8..16+h*8+7 within each 32-K block -> two 16B LDS loads.
__device__ __forceinline__ v16h load_frag(const _Float16* base, int off) {
  v8h lo = *(const v8h*)(base + off);
  v8h hi = *(const v8h*)(base + off + 16);
  v16h a;
#pragma unroll
  for (int i = 0; i < 8; ++i) { a[i] = lo[i]; a[i + 8] = hi[i]; }
  return a;
}

// One MLP layer for one wave's 16-row tile: Y[128xN] = relu(X[128xK] @ W^T + b)
template <int K, int N, bool RELU>
__device__ __forceinline__ void mlp_layer(const _Float16* X, const _Float16* W,
                                          const float* bias, _Float16* Y,
                                          int lane, int m0) {
  const int n = lane & 15;
  const int h = lane >> 4;
  v16h a[K / 32];
#pragma unroll
  for (int kk = 0; kk < K / 32; ++kk)
    a[kk] = load_frag(X + (m0 + n) * K, kk * 32 + h * 8);

#pragma unroll
  for (int nt = 0; nt < N / 16; ++nt) {
    v8f c = {};
#pragma unroll
    for (int kk = 0; kk < K / 32; ++kk) {
      v16h b = load_frag(W + (nt * 16 + n) * K, kk * 32 + h * 8);
      c = __builtin_amdgcn_wmma_f32_16x16x32_f16(false, a[kk], false, b,
                                                 (short)0, c, false, false);
    }
    const float bb = bias[nt * 16 + n];
#pragma unroll
    for (int r = 0; r < 8; ++r) {
      float v = c[r] + bb;
      if (RELU) v = v > 0.f ? v : 0.f;
      Y[(m0 + r + 8 * h) * N + nt * 16 + n] = (_Float16)v;
    }
  }
}

// Hash-encode 8 levels [L0, L0+8) of one point into dst (stride-32 fp16 row).
template <int L0>
__device__ __forceinline__ void encode8(float cx, float cy, float cz,
                                        const _Float16* __restrict__ tables,
                                        _Float16* dst) {
  // floor(16 * 2^(l/3)) for l = 0..15  (coeff = 32^(1/15) = 2^(1/3))
  constexpr float RES[16] = {16.f, 20.f, 25.f, 32.f, 40.f, 50.f, 64.f, 80.f,
                             101.f, 128.f, 161.f, 203.f, 256.f, 322.f, 406.f, 512.f};
#pragma unroll
  for (int li = 0; li < 8; ++li) {
    const float r  = RES[L0 + li];
    const float sx = cx * r, sy = cy * r, sz = cz * r;
    const float fx = floorf(sx), fy = floorf(sy), fz = floorf(sz);
    const float ex = ceilf(sx),  ey = ceilf(sy),  ez = ceilf(sz);
    const int fxi = (int)fx, fyi = (int)fy, fzi = (int)fz;
    const int exi = (int)ex, eyi = (int)ey, ezi = (int)ez;
    const _Float16* tab = tables + (((size_t)(L0 + li)) << 20);  // 2^19 * FEAT
    float a0 = 0.f, a1 = 0.f;
#pragma unroll
    for (int o = 0; o < 8; ++o) {
      const int gx = (o & 4) ? exi : fxi;
      const int gy = (o & 2) ? eyi : fyi;
      const int gz = (o & 1) ? ezi : fzi;
      const float wx = fabsf(((o & 4) ? fx : ex) - sx);
      const float wy = fabsf(((o & 2) ? fy : ey) - sy);
      const float wz = fabsf(((o & 1) ? fz : ez) - sz);
      unsigned idx = ((unsigned)gx) ^ ((unsigned)gy * 2654435761u) ^
                     ((unsigned)gz * 805459861u);
      idx &= (TABLE_SIZE - 1);
      union { unsigned u; _Float16 f[2]; } t;
      t.u = *(const unsigned*)(tab + (size_t)idx * 2);  // one b32 gather (L2-hot)
      const float w = wx * wy * wz;
      a0 += w * (float)t.f[0];
      a1 += w * (float)t.f[1];
    }
    dst[(L0 + li) * 2 + 0] = (_Float16)a0;
    dst[(L0 + li) * 2 + 1] = (_Float16)a1;
  }
}

// One-shot prep: fp32 weights -> fp16 workspace, biases -> f32 workspace.
__global__ __launch_bounds__(256)
void nerf_prep_kernel(const float* __restrict__ w_in,  const float* __restrict__ w_h0,
                      const float* __restrict__ w_h1,  const float* __restrict__ w_out,
                      const float* __restrict__ b_in,  const float* __restrict__ b_h0,
                      const float* __restrict__ b_h1,  const float* __restrict__ b_out,
                      _Float16* __restrict__ wsW, float* __restrict__ wsB) {
  const int i = blockIdx.x * 256 + threadIdx.x;   // grid = 44 -> [0, 11264)
  if (i < W_H0_OFF)        wsW[i] = (_Float16)w_in[i];
  else if (i < W_H1_OFF)   wsW[i] = (_Float16)w_h0[i - W_H0_OFF];
  else if (i < W_OUT_OFF)  wsW[i] = (_Float16)w_h1[i - W_H1_OFF];
  else                     wsW[i] = (_Float16)w_out[i - W_OUT_OFF];
  if (i < B_TOTAL) {
    wsB[i] = (i < 64) ? b_in[i] : (i < 128) ? b_h0[i - 64]
           : (i < 192) ? b_h1[i - 128] : b_out[i - 192];
  }
}

template <bool FROM_WS>
__global__ __launch_bounds__(256)
void nerf_fused_kernel(const float* __restrict__ coords,
                       const _Float16* __restrict__ tables,
                       const float* __restrict__ w_in,  const float* __restrict__ b_in,
                       const float* __restrict__ w_h0,  const float* __restrict__ b_h0,
                       const float* __restrict__ w_h1,  const float* __restrict__ b_h1,
                       const float* __restrict__ w_out, const float* __restrict__ b_out,
                       const _Float16* __restrict__ wsW, const float* __restrict__ wsB,
                       float* __restrict__ out) {
  __shared__ alignas(16) _Float16 sA[PTS_PER_BLK * 64];   // activations ping
  __shared__ alignas(16) _Float16 sB[PTS_PER_BLK * 64];   // activations pong
  __shared__ alignas(16) _Float16 sW[W_TOTAL];            // all fp16 weights
  __shared__ alignas(16) float    sBias[B_TOTAL];

  const int tid = threadIdx.x;

  // --- Stage weights into LDS -------------------------------------------
  if (FROM_WS) {
#if USE_ASYNC_LDS
    // Async DMA straight into LDS (no VGPR round-trip), tracked on ASYNCcnt;
    // issued *before* the gather phase so its latency is fully hidden.
#pragma unroll 2
    for (int i = tid; i < W_TOTAL / 8; i += 256)   // 1408 x 16B
      __builtin_amdgcn_global_load_async_to_lds_b128(V4IP(wsW + i * 8),
                                                     V4IP(&sW[i * 8]), 0, 0);
    if (tid < B_TOTAL / 4)                         // 52 x 16B
      __builtin_amdgcn_global_load_async_to_lds_b128(V4IP(wsB + tid * 4),
                                                     V4IP(&sBias[tid * 4]), 0, 0);
#else
    for (int i = tid; i < W_TOTAL; i += 256) sW[i] = wsW[i];
    if (tid < B_TOTAL) sBias[tid] = wsB[tid];
#endif
  } else {
    // Fallback: convert fp32 weights per block (original path).
    for (int i = tid; i < 2048; i += 256) sW[W_IN_OFF + i]  = (_Float16)w_in[i];
    for (int i = tid; i < 4096; i += 256) sW[W_H0_OFF + i]  = (_Float16)w_h0[i];
    for (int i = tid; i < 4096; i += 256) sW[W_H1_OFF + i]  = (_Float16)w_h1[i];
    for (int i = tid; i < 1024; i += 256) sW[W_OUT_OFF + i] = (_Float16)w_out[i];
    if (tid < 64)        sBias[tid] = b_in[tid];
    else if (tid < 128)  sBias[tid] = b_h0[tid - 64];
    else if (tid < 192)  sBias[tid] = b_h1[tid - 128];
    else if (tid < 208)  sBias[tid] = b_out[tid - 192];
  }

  // --- Hash encoding: 2 threads per point, 8 levels each ------------------
  // Wave w (threads 32w..32w+31) encodes points 16w..16w+15 == the exact
  // rows it consumes in the GEMM phase (no cross-wave RAW after the barrier).
  const int p  = tid >> 1;
  const int gp = blockIdx.x * PTS_PER_BLK + p;
  const float cx = coords[gp * 3 + 0];
  const float cy = coords[gp * 3 + 1];
  const float cz = coords[gp * 3 + 2];
  _Float16* enc = sA + p * 32;
  if (tid & 1) encode8<8>(cx, cy, cz, tables, enc);
  else         encode8<0>(cx, cy, cz, tables, enc);

#if USE_ASYNC_LDS
  if (FROM_WS) {
#if __has_builtin(__builtin_amdgcn_s_wait_asynccnt)
    __builtin_amdgcn_s_wait_asynccnt(0);
#else
    asm volatile("s_wait_asynccnt 0x0" ::: "memory");
#endif
  }
#endif
  __syncthreads();

  // --- WMMA MLP: each wave owns a 16-row M tile ---------------------------
  const int lane = tid & 31;
  const int m0   = (tid >> 5) * 16;

  mlp_layer<32, 64, true>(sA, sW + W_IN_OFF, sBias,       sB, lane, m0);  // 4 wmma
  mlp_layer<64, 64, true>(sB, sW + W_H0_OFF, sBias + 64,  sA, lane, m0);  // 8 wmma
  mlp_layer<64, 64, true>(sA, sW + W_H1_OFF, sBias + 128, sB, lane, m0);  // 8 wmma

  // --- Output layer: [128x64] @ [64x16], f32 straight to HBM --------------
  {
    const int n = lane & 15;
    const int h = lane >> 4;
    v8f c = {};
    v16h a0 = load_frag(sB + (m0 + n) * 64, h * 8);
    v16h b0 = load_frag(sW + W_OUT_OFF + n * 64, h * 8);
    c = __builtin_amdgcn_wmma_f32_16x16x32_f16(false, a0, false, b0, (short)0, c, false, false);
    v16h a1 = load_frag(sB + (m0 + n) * 64, 32 + h * 8);
    v16h b1 = load_frag(sW + W_OUT_OFF + n * 64, 32 + h * 8);
    c = __builtin_amdgcn_wmma_f32_16x16x32_f16(false, a1, false, b1, (short)0, c, false, false);

    const float bb = sBias[192 + n];
    const size_t base = (size_t)blockIdx.x * PTS_PER_BLK + m0;
#pragma unroll
    for (int r = 0; r < 8; ++r)
      out[(base + r + 8 * h) * 16 + n] = c[r] + bb;
  }
}

extern "C" void kernel_launch(void* const* d_in, const int* in_sizes, int n_in,
                              void* d_out, int out_size, void* d_ws, size_t ws_size,
                              hipStream_t stream) {
  (void)in_sizes; (void)n_in; (void)out_size;
  const float*    coords = (const float*)d_in[0];
  const _Float16* tables = (const _Float16*)d_in[1];
  const float*    w_in   = (const float*)d_in[2];
  const float*    b_in   = (const float*)d_in[3];
  const float*    w_h0   = (const float*)d_in[4];
  const float*    b_h0   = (const float*)d_in[5];
  const float*    w_h1   = (const float*)d_in[6];
  const float*    b_h1   = (const float*)d_in[7];
  const float*    w_out  = (const float*)d_in[8];
  const float*    b_out  = (const float*)d_in[9];
  float*          out    = (float*)d_out;

  const size_t WS_W_BYTES = (size_t)W_TOTAL * sizeof(_Float16);   // 22528
  const size_t WS_BYTES   = WS_W_BYTES + (size_t)B_TOTAL * sizeof(float);

  dim3 grid(N_POINTS / PTS_PER_BLK);  // 2048 blocks of 256 (8 wave32)
  dim3 block(256);

  if (ws_size >= WS_BYTES) {
    _Float16* wsW = (_Float16*)d_ws;
    float*    wsB = (float*)((char*)d_ws + WS_W_BYTES);
    nerf_prep_kernel<<<dim3(W_TOTAL / 256), block, 0, stream>>>(
        w_in, w_h0, w_h1, w_out, b_in, b_h0, b_h1, b_out, wsW, wsB);
    nerf_fused_kernel<true><<<grid, block, 0, stream>>>(
        coords, tables, w_in, b_in, w_h0, b_h0, w_h1, b_h1, w_out, b_out,
        wsW, wsB, out);
  } else {
    nerf_fused_kernel<false><<<grid, block, 0, stream>>>(
        coords, tables, w_in, b_in, w_h0, b_h0, w_h1, b_h1, w_out, b_out,
        nullptr, nullptr, out);
  }
}
